// BasicHypergraphConv_67010079752614
// MI455X (gfx1250) — matrix operand order repaired and checked by reading, not compiled, not run on hardware
//
#include <hip/hip_runtime.h>

// ---------------- problem constants (match reference) ----------------
#define NNODES 50000
#define NEDGES 10000
#define NCONN  150000
#define DIM    512
#define SCAN_B 256

typedef __attribute__((ext_vector_type(16))) __bf16 v16bf;
typedef __attribute__((ext_vector_type(8)))  float  v8f;

// fp32 -> bf16 (round to nearest even), packed pair into one dword
__device__ __forceinline__ unsigned f2bf_pk(float a, float b) {
    unsigned ua = __float_as_uint(a);
    unsigned ub = __float_as_uint(b);
    ua = (ua + 0x7FFFu + ((ua >> 16) & 1u)) >> 16;
    ub = (ub + 0x7FFFu + ((ub >> 16) & 1u)) >> 16;
    return ua | (ub << 16);
}

union BF16Frag {
    v16bf    bf;
    unsigned u[8];
    uint4    q[2];
};

// ---------------- fp32 -> bf16 streaming conversion (8 floats / thread) ----------------
__global__ void cvt_bf16_kernel(const float* __restrict__ src, unsigned* __restrict__ dst, int nq) {
    int i = blockIdx.x * blockDim.x + threadIdx.x;   // one uint4 (8 bf16) per thread
    if (i < nq) {
        const float4* p = (const float4*)src + 2 * (size_t)i;
        float4 f0 = p[0], f1 = p[1];
        uint4 o;
        o.x = f2bf_pk(f0.x, f0.y);
        o.y = f2bf_pk(f0.z, f0.w);
        o.z = f2bf_pk(f1.x, f1.y);
        o.w = f2bf_pk(f1.z, f1.w);
        ((uint4*)dst)[i] = o;
    }
}

// ---------------- zero int fill ----------------
__global__ void zero_i32_kernel(int* __restrict__ p, int n) {
    int i = blockIdx.x * blockDim.x + threadIdx.x;
    int stride = gridDim.x * blockDim.x;
    for (; i < n; i += stride) p[i] = 0;
}

// ---------------- h = x @ W^T + b via v_wmma_f32_16x16x32_bf16 ----------------
// block = 256 threads (8 waves). Block covers 32 rows x 512 cols.
// Wave w covers cols [w*64, w*64+64): acc[2 Mtiles][4 Ntiles]. Both A and B are
// pre-converted bf16, so the K-loop is pure b128 loads + WMMAs (no VALU packing).
__global__ __launch_bounds__(256)
void gemm_kernel(const unsigned* __restrict__ Xb, const unsigned* __restrict__ Wb,
                 const float* __restrict__ bias, float* __restrict__ h) {
    const int m0   = blockIdx.x * 32;
    const int wave = threadIdx.x >> 5;
    const int lane = threadIdx.x & 31;
    const int idx  = lane & 15;
    const int half = lane >> 4;

    v8f acc[2][4];
    v8f zero = {};
#pragma unroll
    for (int mt = 0; mt < 2; ++mt)
#pragma unroll
        for (int nt = 0; nt < 4; ++nt) acc[mt][nt] = zero;

    // A rows (clamped in the tail block; EXEC stays all-ones for WMMA).
    // A frag per lane: K in [k0+half*8, +8) and [k0+16+half*8, +8)  (2 x 16B)
    int r0 = m0 + idx;      if (r0 > NNODES - 1) r0 = NNODES - 1;
    int r1 = m0 + 16 + idx; if (r1 > NNODES - 1) r1 = NNODES - 1;
    const unsigned* xA0 = Xb + ((r0 * DIM + half * 8) >> 1);
    const unsigned* xA1 = Xb + ((r1 * DIM + half * 8) >> 1);
    // B frag per lane: col (wave*64 + nt*16 + idx) = row of W; K in
    // [k0+half*8, +8) and [k0+16+half*8, +8)  (2 x 16B, same as A)
    const unsigned* wB = Wb + (((wave * 64 + idx) * DIM + half * 8) >> 1);

    for (int k0 = 0; k0 < DIM; k0 += 32) {
        // preload all 4 B fragments so vmem latency overlaps the WMMA chain
        BF16Frag bf[4];
#pragma unroll
        for (int nt = 0; nt < 4; ++nt) {
            const unsigned* wr = wB + ((nt * 16 * DIM + k0) >> 1);
            bf[nt].q[0] = *(const uint4*)(wr);
            bf[nt].q[1] = *(const uint4*)(wr + 8);
        }
        BF16Frag a0, a1;
        a0.q[0] = *(const uint4*)(xA0 + (k0 >> 1));
        a0.q[1] = *(const uint4*)(xA0 + ((k0 + 16) >> 1));
        a1.q[0] = *(const uint4*)(xA1 + (k0 >> 1));
        a1.q[1] = *(const uint4*)(xA1 + ((k0 + 16) >> 1));
#pragma unroll
        for (int nt = 0; nt < 4; ++nt) {
            acc[0][nt] = __builtin_amdgcn_wmma_f32_16x16x32_bf16(
                false, a0.bf, false, bf[nt].bf, (short)0, acc[0][nt], false, false);
            acc[1][nt] = __builtin_amdgcn_wmma_f32_16x16x32_bf16(
                false, a1.bf, false, bf[nt].bf, (short)0, acc[1][nt], false, false);
        }
    }

    // C/D layout: VGPR j -> M = j + half*8; N = lane%16.
    // Block-uniform fast path: only the last block can have rows >= NNODES.
    if (m0 + 32 <= NNODES) {
#pragma unroll
        for (int mt = 0; mt < 2; ++mt)
#pragma unroll
            for (int nt = 0; nt < 4; ++nt) {
                const int n0 = wave * 64 + nt * 16;
                const float bv = bias[n0 + idx];
#pragma unroll
                for (int j = 0; j < 8; ++j) {
                    const int row = m0 + mt * 16 + half * 8 + j;
                    h[row * DIM + n0 + idx] = acc[mt][nt][j] + bv;
                }
            }
    } else {
#pragma unroll
        for (int mt = 0; mt < 2; ++mt)
#pragma unroll
            for (int nt = 0; nt < 4; ++nt) {
                const int n0 = wave * 64 + nt * 16;
                const float bv = bias[n0 + idx];
#pragma unroll
                for (int j = 0; j < 8; ++j) {
                    const int row = m0 + mt * 16 + half * 8 + j;
                    if (row < NNODES) h[row * DIM + n0 + idx] = acc[mt][nt][j] + bv;
                }
            }
    }
}

// ---------------- CSR build: histogram ----------------
__global__ void hist_kernel(const int* __restrict__ nodes, const int* __restrict__ edges,
                            int* __restrict__ ecnt, int* __restrict__ ncnt) {
    int i = blockIdx.x * blockDim.x + threadIdx.x;
    if (i < NCONN) {
        atomicAdd(ecnt + edges[i], 1);
        atomicAdd(ncnt + nodes[i], 1);
    }
}

// ---------------- CSR build: 3-kernel exclusive scan (n <= 65536) ----------------
__global__ void scan1_kernel(const int* __restrict__ in, int* __restrict__ out,
                             int* __restrict__ bsums, int n) {
    __shared__ int tmp[SCAN_B];
    int g = blockIdx.x * SCAN_B + threadIdx.x;
    int v = (g < n) ? in[g] : 0;
    tmp[threadIdx.x] = v;
    __syncthreads();
    for (int off = 1; off < SCAN_B; off <<= 1) {
        int t = (threadIdx.x >= off) ? tmp[threadIdx.x - off] : 0;
        __syncthreads();
        tmp[threadIdx.x] += t;
        __syncthreads();
    }
    if (g < n) out[g] = tmp[threadIdx.x] - v;              // exclusive
    if (threadIdx.x == SCAN_B - 1) bsums[blockIdx.x] = tmp[threadIdx.x];
}
__global__ void scan2_kernel(int* __restrict__ bsums, int nb) {
    __shared__ int tmp[SCAN_B];
    int v = (threadIdx.x < nb) ? bsums[threadIdx.x] : 0;
    tmp[threadIdx.x] = v;
    __syncthreads();
    for (int off = 1; off < SCAN_B; off <<= 1) {
        int t = (threadIdx.x >= off) ? tmp[threadIdx.x - off] : 0;
        __syncthreads();
        tmp[threadIdx.x] += t;
        __syncthreads();
    }
    if (threadIdx.x < nb) bsums[threadIdx.x] = tmp[threadIdx.x] - v;  // exclusive
}
__global__ void scan3_kernel(int* __restrict__ out, const int* __restrict__ bsums, int n) {
    int g = blockIdx.x * SCAN_B + threadIdx.x;
    if (g < n) out[g] += bsums[blockIdx.x];
}

// ---------------- CSR build: fill member lists ----------------
__global__ void fill_kernel(const int* __restrict__ nodes, const int* __restrict__ edges,
                            const int* __restrict__ estart, int* __restrict__ ecur,
                            int* __restrict__ eidx,
                            const int* __restrict__ nstart, int* __restrict__ ncur,
                            int* __restrict__ nidx) {
    int i = blockIdx.x * blockDim.x + threadIdx.x;
    if (i < NCONN) {
        int e = edges[i];
        int se = atomicAdd(ecur + e, 1);
        eidx[estart[e] + se] = i;
        int n = nodes[i];
        int sn = atomicAdd(ncur + n, 1);
        nidx[nstart[n] + sn] = i;
    }
}

// ---------------- node -> edge gather-mean (no atomics; one block per edge) ----------
__global__ __launch_bounds__(256)
void edge_gather_kernel(const float* __restrict__ h, const int* __restrict__ nodes,
                        const int* __restrict__ eidx, const int* __restrict__ estart,
                        const int* __restrict__ ecnt, float* __restrict__ ef) {
    const int e = blockIdx.x;
    const int c = threadIdx.x;          // owns cols c and c+256
    const int s = estart[e];
    const int deg = ecnt[e];
    float a0 = 0.0f, a1 = 0.0f;
    for (int j = 0; j < deg; ++j) {
        const float* hr = h + nodes[eidx[s + j]] * DIM;
        a0 += hr[c];
        a1 += hr[c + 256];
    }
    const float inv = 1.0f / fmaxf((float)deg, 1.0f);
    ef[e * DIM + c]       = a0 * inv;
    ef[e * DIM + c + 256] = a1 * inv;
}

// ---------------- edge -> node gather-mean (no atomics; one block per node) ----------
__global__ __launch_bounds__(256)
void node_gather_kernel(const float* __restrict__ ef, const int* __restrict__ edges,
                        const int* __restrict__ nidx, const int* __restrict__ nstart,
                        const int* __restrict__ ncnt, float* __restrict__ out) {
    const int n = blockIdx.x;
    const int c = threadIdx.x;
    const int s = nstart[n];
    const int deg = ncnt[n];
    float a0 = 0.0f, a1 = 0.0f;
    for (int j = 0; j < deg; ++j) {
        const float* er = ef + edges[nidx[s + j]] * DIM;
        a0 += er[c];
        a1 += er[c + 256];
    }
    const float inv = 1.0f / fmaxf((float)deg, 1.0f);
    out[n * DIM + c]       = a0 * inv;
    out[n * DIM + c + 256] = a1 * inv;
}

// ---------------- launcher ----------------
extern "C" void kernel_launch(void* const* d_in, const int* in_sizes, int n_in,
                              void* d_out, int out_size, void* d_ws, size_t ws_size,
                              hipStream_t stream) {
    (void)in_sizes; (void)n_in; (void)out_size; (void)ws_size;

    const float* x     = (const float*)d_in[0];   // [50000, 512]
    const float* W     = (const float*)d_in[1];   // [512, 512]
    const float* bias  = (const float*)d_in[2];   // [512]
    const int*   nodes = (const int*)d_in[3];     // [150000]
    const int*   edges = (const int*)d_in[4];     // [150000]
    float*       out   = (float*)d_out;           // [50000, 512]

    // workspace layout (256B-aligned segments)
    char* ws = (char*)d_ws;
    size_t ofs = 0;
    auto carve = [&](size_t bytes) { size_t o = ofs; ofs += (bytes + 255) & ~(size_t)255; return o; };
    float*    h      = (float*)   (ws + carve((size_t)NNODES * DIM * 4)); // 102.4 MB
    float*    ef     = (float*)   (ws + carve((size_t)NEDGES * DIM * 4)); // 20.5 MB
    unsigned* Xb     = (unsigned*)(ws + carve((size_t)NNODES * DIM * 2)); // 51.2 MB (x in bf16)
    unsigned* Wb     = (unsigned*)(ws + carve((size_t)DIM * DIM * 2));    // 512 KB
    size_t    zbase  = ofs;                                               // zeroed region start
    int*      ecnt   = (int*)(ws + carve(NEDGES * 4));
    int*      ecur   = (int*)(ws + carve(NEDGES * 4));
    int*      ncnt   = (int*)(ws + carve(NNODES * 4));
    int*      ncur   = (int*)(ws + carve(NNODES * 4));
    size_t    zend   = ofs;                                               // zeroed region end
    int*      estart = (int*)(ws + carve(NEDGES * 4));
    int*      nstart = (int*)(ws + carve(NNODES * 4));
    int*      ebs    = (int*)(ws + carve(SCAN_B * 4));
    int*      nbs    = (int*)(ws + carve(SCAN_B * 4));
    int*      eidx   = (int*)(ws + carve(NCONN * 4));
    int*      nidx   = (int*)(ws + carve(NCONN * 4));

    const int enb = (NEDGES + SCAN_B - 1) / SCAN_B;   // 40
    const int nnb = (NNODES + SCAN_B - 1) / SCAN_B;   // 196

    // 1) x and W -> bf16 (single-pass streaming; removes all packing VALU from GEMM)
    {
        int nqx = NNODES * DIM / 8;   // 3,200,000 uint4s
        cvt_bf16_kernel<<<(nqx + 255) / 256, 256, 0, stream>>>(x, Xb, nqx);
        int nqw = DIM * DIM / 8;      // 32,768 uint4s
        cvt_bf16_kernel<<<(nqw + 255) / 256, 256, 0, stream>>>(W, Wb, nqw);
    }
    // 2) zero histogram/cursor region (workspace is not re-poisoned between replays)
    {
        int nz = (int)((zend - zbase) / 4);
        zero_i32_kernel<<<128, 256, 0, stream>>>((int*)(ws + zbase), nz);
    }
    // 3) GEMM (WMMA bf16, fp32 accumulate)
    gemm_kernel<<<(NNODES + 31) / 32, 256, 0, stream>>>(Xb, Wb, bias, h);
    // 4) CSR build: histogram -> exclusive scans -> fill
    hist_kernel<<<(NCONN + 255) / 256, 256, 0, stream>>>(nodes, edges, ecnt, ncnt);
    scan1_kernel<<<enb, SCAN_B, 0, stream>>>(ecnt, estart, ebs, NEDGES);
    scan2_kernel<<<1, SCAN_B, 0, stream>>>(ebs, enb);
    scan3_kernel<<<enb, SCAN_B, 0, stream>>>(estart, ebs, NEDGES);
    scan1_kernel<<<nnb, SCAN_B, 0, stream>>>(ncnt, nstart, nbs, NNODES);
    scan2_kernel<<<1, SCAN_B, 0, stream>>>(nbs, nnb);
    scan3_kernel<<<nnb, SCAN_B, 0, stream>>>(nstart, nbs, NNODES);
    fill_kernel<<<(NCONN + 255) / 256, 256, 0, stream>>>(nodes, edges, estart, ecur, eidx,
                                                         nstart, ncur, nidx);
    // 5) gather-mean stages (atomic-free, L2-resident streaming)
    edge_gather_kernel<<<NEDGES, 256, 0, stream>>>(h, nodes, eidx, estart, ecnt, ef);
    node_gather_kernel<<<NNODES, 256, 0, stream>>>(ef, edges, nidx, nstart, ncnt, out);
}
